// GraphACModel_46454366273716
// MI455X (gfx1250) — matrix operand order repaired and checked by reading, not compiled, not run on hardware
//
#include <hip/hip_runtime.h>
#include <hip/hip_bf16.h>
#include <math.h>

#define N_REQ 500000
#define N_VEH 100000
#define N_PAS 200000
#define NB    4096
#define E_V2P 1000000
#define E_R2V 4000000

typedef __attribute__((ext_vector_type(16))) _Float16 v16h;
typedef __attribute__((ext_vector_type(8)))  float    v8f;

// Branch-free fast tanh: 1 - 2/(e^{2x}+1). Uses v_exp_f32 + v_rcp_f32,
// saturates to +/-1 for large |x|; keeps EXEC uniform around WMMA sites.
__device__ __forceinline__ float tanh_fast(float x) {
    float e = __expf(2.0f * x);
    return 1.0f - __fdividef(2.0f, e + 1.0f);
}

// ---------- order-preserving float <-> uint encoding for atomicMax ----------
__device__ __forceinline__ unsigned fenc(float f) {
    unsigned u = __float_as_uint(f);
    return (u & 0x80000000u) ? ~u : (u | 0x80000000u);
}
__device__ __forceinline__ float fdec(unsigned e) {
    return (e & 0x80000000u) ? __uint_as_float(e & 0x7fffffffu) : __uint_as_float(~e);
}

// ======================= WMMA MLP building blocks ==========================
// A fragment: 16x32 fp16 tile slice from a row-major [16][64] fp16 LDS tile.
// ISA layout: lanes 0-15 hold M=lane, K=kbase+{0..7} in v0-3 and K=kbase+{16..23}
// in v4-7; lanes 16-31 hold K offsets +8.
__device__ __forceinline__ v16h load_afrag(const _Float16* lds, int lane, int kbase) {
    int m  = lane & 15;
    int lo = (lane & 16) ? 8 : 0;
    const _Float16* p0 = lds + m * 64 + kbase + lo;
    const _Float16* p1 = p0 + 16;
    v16h r;
#pragma unroll
    for (int j = 0; j < 8; ++j) { r[j] = p0[j]; r[j + 8] = p1[j]; }
    return r;
}

// B fragment: weights stored column-major fp16, wt[n*64 + k].
// Lane n (0-15): col n, K=kstep*32+{0..7,16..23}; lane n+16: col n, K=kstep*32+{8..15,24..31}.
__device__ __forceinline__ v16h load_bfrag(const _Float16* wt, int lane, int ntile, int kstep) {
    const _Float16* p = wt + (ntile * 16 + (lane & 15)) * 64 + kstep * 32 + ((lane & 16) ? 8 : 0);
    v16h r;
#pragma unroll
    for (int j = 0; j < 8; ++j) { r[j] = p[j]; r[j + 8] = p[j + 16]; }
    return r;
}

// One 16x64 -> 16x64 layer: D = tanh(A @ W + b), A read from LDS tile,
// result written back into the same LDS tile (A-staging format for next layer).
__device__ __forceinline__ void wmma_layer(_Float16* lds, const _Float16* wt,
                                           const float* bias, int lane) {
    v16h a0 = load_afrag(lds, lane, 0);
    v16h a1 = load_afrag(lds, lane, 32);
    __syncthreads();
    int mofs = (lane & 16) ? 8 : 0;
#pragma unroll
    for (int nt = 0; nt < 4; ++nt) {
        v16h b0 = load_bfrag(wt, lane, nt, 0);
        v16h b1 = load_bfrag(wt, lane, nt, 1);
        v8f acc = {0.f, 0.f, 0.f, 0.f, 0.f, 0.f, 0.f, 0.f};
        acc = __builtin_amdgcn_wmma_f32_16x16x32_f16(false, a0, false, b0, (short)0, acc, false, false);
        acc = __builtin_amdgcn_wmma_f32_16x16x32_f16(false, a1, false, b1, (short)0, acc, false, false);
        int n = nt * 16 + (lane & 15);
        float bv = bias[n];
#pragma unroll
        for (int j = 0; j < 8; ++j) {
            float v = tanh_fast(acc[j] + bv);
            lds[(j + mofs) * 64 + n] = (_Float16)v;   // C/D layout: M=j+mofs, N=n
        }
    }
    __syncthreads();
}

// ============================ utility kernels ==============================
__global__ void k_zero(float4* p, int n4) {
    int i = blockIdx.x * 256 + threadIdx.x;
    if (i < n4) p[i] = make_float4(0.f, 0.f, 0.f, 0.f);
}

__global__ void k_initmax(unsigned* bmax) {
    int i = blockIdx.x * 256 + threadIdx.x;
    if (i < NB) bmax[i] = fenc(-3.402823e38f);
}

// Transpose + fp16-convert the four 64-wide weight matrices (K padded to 64).
__global__ void k_wconv(const float* w1, const float* w2, const float* c1, const float* c2,
                        _Float16* w1t, _Float16* w2t, _Float16* c1t, _Float16* c2t) {
    int i = blockIdx.x * 256 + threadIdx.x;   // 0 .. 16383
    if (i >= 4 * 4096) return;
    int tbl = i >> 12, e = i & 4095;
    int n = e >> 6, k = e & 63;
    if (tbl == 0)      w1t[n * 64 + k] = (k < 48) ? (_Float16)w1[k * 64 + n] : (_Float16)0.0f;
    else if (tbl == 1) w2t[n * 64 + k] = (_Float16)w2[k * 64 + n];
    else if (tbl == 2) c1t[n * 64 + k] = (_Float16)c1[k * 64 + n];
    else               c2t[n * 64 + k] = (_Float16)c2[k * 64 + n];
}

// ============================ node encoders ================================
__global__ void k_req_encode(const float* rx, const int* rbatch,
                             const float* aw, const float* ab,
                             const float* cw, const float* cb,
                             _Float16* reqf16, float* reqmean, float* reqcnt) {
    __shared__ float s[352];   // aw 160 | ab 16 | cw 160 | cb 16
    int tid = threadIdx.x;
    for (int j = tid; j < 352; j += 256)
        s[j] = (j < 160) ? aw[j] : (j < 176) ? ab[j - 160] : (j < 336) ? cw[j - 176] : cb[j - 336];
    __syncthreads();
    int i = blockIdx.x * 256 + tid;
    if (i >= N_REQ) return;
    float x[10];
#pragma unroll
    for (int j = 0; j < 10; ++j) x[j] = rx[i * 10 + j];
    int b = rbatch[i];
#pragma unroll
    for (int c = 0; c < 16; ++c) {
        float fa = s[160 + c], fc = s[336 + c];
#pragma unroll
        for (int j = 0; j < 10; ++j) { fa += x[j] * s[j * 16 + c]; fc += x[j] * s[176 + j * 16 + c]; }
        reqf16[i * 16 + c] = (_Float16)tanh_fast(fa);
        atomicAdd(&reqmean[b * 16 + c], tanh_fast(fc));
    }
    atomicAdd(&reqcnt[b], 1.0f);
}

__global__ void k_veh_encode(const float* vx, const int* vbatch,
                             const float* aw, const float* ab,
                             const float* cw, const float* cb,
                             float* vehcat, float* cvehf, float* vehmean, float* vehcnt) {
    __shared__ float s[288];   // aw 128 | ab 16 | cw 128 | cb 16
    int tid = threadIdx.x;
    for (int j = tid; j < 288; j += 256)
        s[j] = (j < 128) ? aw[j] : (j < 144) ? ab[j - 128] : (j < 272) ? cw[j - 144] : cb[j - 272];
    __syncthreads();
    int i = blockIdx.x * 256 + tid;
    if (i >= N_VEH) return;
    float x[8];
#pragma unroll
    for (int j = 0; j < 8; ++j) x[j] = vx[i * 8 + j];
    int b = vbatch[i];
#pragma unroll
    for (int c = 0; c < 16; ++c) {
        float fa = s[128 + c], fc = s[272 + c];
#pragma unroll
        for (int j = 0; j < 8; ++j) { fa += x[j] * s[j * 16 + c]; fc += x[j] * s[144 + j * 16 + c]; }
        vehcat[i * 32 + c] = tanh_fast(fa);
        float t = tanh_fast(fc);
        cvehf[i * 16 + c] = t;
        atomicAdd(&vehmean[b * 16 + c], t);
    }
    atomicAdd(&vehcnt[b], 1.0f);
}

// veh2pas: encode passenger on the fly per edge, scatter-sum onto the vehicle.
__global__ void k_v2p(const float* px, const int* es, const int* er,
                      const float* aw, const float* ab, const float* cw, const float* cb,
                      float* passum_a, float* passum_c, float* pascnt) {
    __shared__ float s[352];
    int tid = threadIdx.x;
    for (int j = tid; j < 352; j += 256)
        s[j] = (j < 160) ? aw[j] : (j < 176) ? ab[j - 160] : (j < 336) ? cw[j - 176] : cb[j - 336];
    __syncthreads();
    int e = blockIdx.x * 256 + tid;
    if (e >= E_V2P) return;
    int veh = es[e], pas = er[e];
    float x[10];
#pragma unroll
    for (int j = 0; j < 10; ++j) x[j] = px[pas * 10 + j];
#pragma unroll
    for (int c = 0; c < 16; ++c) {
        float fa = s[160 + c], fc = s[336 + c];
#pragma unroll
        for (int j = 0; j < 10; ++j) { fa += x[j] * s[j * 16 + c]; fc += x[j] * s[176 + j * 16 + c]; }
        atomicAdd(&passum_a[veh * 16 + c], tanh_fast(fa));
        atomicAdd(&passum_c[veh * 16 + c], tanh_fast(fc));
    }
    atomicAdd(&pascnt[veh], 1.0f);
}

__global__ void k_pasfin(const float* passum_a, const float* passum_c, const float* pascnt,
                         float* vehcat, float* cpasmean) {
    int i = blockIdx.x * 256 + threadIdx.x;
    if (i >= N_VEH) return;
    float inv = __fdividef(1.0f, fmaxf(pascnt[i], 1.0f));
#pragma unroll
    for (int c = 0; c < 16; ++c) {
        vehcat[i * 32 + 16 + c] = passum_a[i * 16 + c] * inv;
        cpasmean[i * 16 + c]    = passum_c[i * 16 + c] * inv;
    }
}

// req2veh: the heavy 4M-edge scatter (accumulator fits in L2).
__global__ void k_r2v(const int* es, const int* er, const float* vehcat,
                      float* aggsum, float* aggcnt) {
    int e = blockIdx.x * 256 + threadIdx.x;
    if (e >= E_R2V) return;
    int s = es[e], r = er[e];
#pragma unroll
    for (int c = 0; c < 32; ++c) atomicAdd(&aggsum[s * 32 + c], vehcat[r * 32 + c]);
    atomicAdd(&aggcnt[s], 1.0f);
}

// ======================== actor MLP (WMMA) =================================
__global__ __launch_bounds__(32) void k_actor_mlp(
        const _Float16* reqf16, const float* aggsum, const float* aggcnt,
        const _Float16* w1t, const float* b1, const _Float16* w2t, const float* b2,
        const float* w3, const float* b3, const int* rbatch,
        float* logits, unsigned* bmax) {
    __shared__ _Float16 lds[16 * 64];
    int lane = threadIdx.x;
    int row0 = blockIdx.x * 16;
    // Stage features: [req_f(16) | veh_agg mean(32) | zero pad(16)] as fp16.
    for (int idx = lane; idx < 16 * 64; idx += 32) {
        int m = idx >> 6, k = idx & 63;
        int r = row0 + m;
        float v;
        if (k < 16)      v = (float)reqf16[r * 16 + k];
        else if (k < 48) v = aggsum[r * 32 + (k - 16)] * __fdividef(1.0f, fmaxf(aggcnt[r], 1.0f));
        else             v = 0.0f;
        lds[idx] = (_Float16)v;
    }
    __syncthreads();
    wmma_layer(lds, w1t, b1, lane);
    wmma_layer(lds, w2t, b2, lane);
    if (lane < 16) {
        float acc = b3[0];
        for (int k = 0; k < 64; ++k) acc += (float)lds[lane * 64 + k] * w3[k];
        int r = row0 + lane;
        logits[r] = acc;
        atomicMax(&bmax[rbatch[r]], fenc(acc));
    }
}

// ======================== softmax / entropy passes =========================
__global__ void k_sm2(const float* logits, const int* rbatch, const unsigned* bmax, float* bsum) {
    int i = blockIdx.x * 256 + threadIdx.x;
    if (i >= N_REQ) return;
    int b = rbatch[i];
    atomicAdd(&bsum[b], __expf(logits[i] - fdec(bmax[b])));
}

__global__ void k_sm3(const float* logits, const int* rbatch, const unsigned* bmax,
                      const float* bsum, float* bent) {
    int i = blockIdx.x * 256 + threadIdx.x;
    if (i >= N_REQ) return;
    int b = rbatch[i];
    float lp = logits[i] - fdec(bmax[b]) - __logf(bsum[b]);
    atomicAdd(&bent[b], -__expf(lp) * lp);
}

// ======================== critic MLP (WMMA) ================================
__global__ __launch_bounds__(32) void k_critic(
        const float* reqmean, const float* reqcnt,
        const float* cvehf, const float* cpasmean,
        const float* vehmean, const float* vehcnt, const int* start_idx,
        const _Float16* c1t, const float* cb1, const _Float16* c2t, const float* cb2,
        const float* cw3, const float* cb3, float* out_value) {
    __shared__ _Float16 lds[16 * 64];
    int lane = threadIdx.x;
    int row0 = blockIdx.x * 16;
    // Features: [req_mean(16) | c_veh_f[start](16) | c_pas_mean[start](16) | veh_mean(16)]
    for (int idx = lane; idx < 16 * 64; idx += 32) {
        int m = idx >> 6, k = idx & 63, c = k & 15;
        int g = row0 + m;
        float v;
        if (k < 16)      v = reqmean[g * 16 + c] * __fdividef(1.0f, fmaxf(reqcnt[g], 1.0f));
        else if (k < 32) v = cvehf[start_idx[g] * 16 + c];
        else if (k < 48) v = cpasmean[start_idx[g] * 16 + c];
        else             v = vehmean[g * 16 + c] * __fdividef(1.0f, fmaxf(vehcnt[g], 1.0f));
        lds[idx] = (_Float16)v;
    }
    __syncthreads();
    wmma_layer(lds, c1t, cb1, lane);
    wmma_layer(lds, c2t, cb2, lane);
    if (lane < 16) {
        float acc = cb3[0];
        for (int k = 0; k < 64; ++k) acc += (float)lds[lane * 64 + k] * cw3[k];
        out_value[row0 + lane] = acc;
    }
}

// ==================== action log-probs + entropy mean ======================
__global__ void k_final(const float* logits, const int* rbatch, const unsigned* bmax,
                        const float* bsum, const int* action_idx, const float* bent, float* out) {
    int i = blockIdx.x * 256 + threadIdx.x;
    if (i < NB) {
        int a = action_idx[i];
        int b = rbatch[a];
        out[NB + i] = logits[a] - fdec(bmax[b]) - __logf(bsum[b]);
    }
    if (blockIdx.x == 0) {
        __shared__ float red[256];
        float acc = 0.f;
        for (int j = threadIdx.x; j < NB; j += 256) acc += bent[j];
        red[threadIdx.x] = acc;
        __syncthreads();
        for (int st = 128; st > 0; st >>= 1) {
            if (threadIdx.x < st) red[threadIdx.x] += red[threadIdx.x + st];
            __syncthreads();
        }
        if (threadIdx.x == 0) out[2 * NB] = red[0] / (float)NB;
    }
}

// =============================== launcher ==================================
extern "C" void kernel_launch(void* const* d_in, const int* in_sizes, int n_in,
                              void* d_out, int out_size, void* d_ws, size_t ws_size,
                              hipStream_t stream) {
    (void)in_sizes; (void)n_in; (void)out_size; (void)ws_size;
    const float* requests_x   = (const float*)d_in[0];
    const float* vehicles_x   = (const float*)d_in[1];
    const float* passengers_x = (const float*)d_in[2];
    const int*   rbatch       = (const int*)d_in[3];
    const int*   vbatch       = (const int*)d_in[4];
    const int*   v2p_s        = (const int*)d_in[5];
    const int*   v2p_r        = (const int*)d_in[6];
    const int*   r2v_s        = (const int*)d_in[7];
    const int*   r2v_r        = (const int*)d_in[8];
    const int*   start_idx    = (const int*)d_in[9];
    const int*   action_idx   = (const int*)d_in[10];
    // params: (w,b) pairs in dict order
    const float* a_req_w = (const float*)d_in[11];  const float* a_req_b = (const float*)d_in[12];
    const float* a_veh_w = (const float*)d_in[13];  const float* a_veh_b = (const float*)d_in[14];
    const float* a_pas_w = (const float*)d_in[15];  const float* a_pas_b = (const float*)d_in[16];
    const float* a_l1_w  = (const float*)d_in[17];  const float* a_l1_b  = (const float*)d_in[18];
    const float* a_l2_w  = (const float*)d_in[19];  const float* a_l2_b  = (const float*)d_in[20];
    const float* a_l3_w  = (const float*)d_in[21];  const float* a_l3_b  = (const float*)d_in[22];
    const float* c_req_w = (const float*)d_in[23];  const float* c_req_b = (const float*)d_in[24];
    const float* c_veh_w = (const float*)d_in[25];  const float* c_veh_b = (const float*)d_in[26];
    const float* c_pas_w = (const float*)d_in[27];  const float* c_pas_b = (const float*)d_in[28];
    const float* c_l1_w  = (const float*)d_in[29];  const float* c_l1_b  = (const float*)d_in[30];
    const float* c_l2_w  = (const float*)d_in[31];  const float* c_l2_b  = (const float*)d_in[32];
    const float* c_l3_w  = (const float*)d_in[33];  const float* c_l3_b  = (const float*)d_in[34];
    float* out = (float*)d_out;

    // ---- workspace carve (256B aligned) ----
    char* base = (char*)d_ws;
    size_t off = 0;
    auto carve = [&](size_t bytes) -> void* {
        off = (off + 255) & ~(size_t)255;
        void* p = base + off;
        off += bytes;
        return p;
    };
    _Float16* reqf16   = (_Float16*)carve((size_t)N_REQ * 16 * 2);
    float*    aggsum   = (float*)carve((size_t)N_REQ * 32 * 4);
    float*    aggcnt   = (float*)carve((size_t)N_REQ * 4);
    float*    logits   = (float*)carve((size_t)N_REQ * 4);
    float*    passum_a = (float*)carve((size_t)N_VEH * 16 * 4);
    float*    passum_c = (float*)carve((size_t)N_VEH * 16 * 4);
    float*    pascnt   = (float*)carve((size_t)N_VEH * 4);
    float*    vehcat   = (float*)carve((size_t)N_VEH * 32 * 4);
    float*    cvehf    = (float*)carve((size_t)N_VEH * 16 * 4);
    float*    cpasmean = (float*)carve((size_t)N_VEH * 16 * 4);
    float*    reqmean  = (float*)carve((size_t)NB * 16 * 4);
    float*    reqcnt   = (float*)carve((size_t)NB * 4);
    float*    vehmean  = (float*)carve((size_t)NB * 16 * 4);
    float*    vehcnt   = (float*)carve((size_t)NB * 4);
    unsigned* bmax     = (unsigned*)carve((size_t)NB * 4);
    float*    bsum     = (float*)carve((size_t)NB * 4);
    float*    bent     = (float*)carve((size_t)NB * 4);
    _Float16* w1t      = (_Float16*)carve(4096 * 2);
    _Float16* w2t      = (_Float16*)carve(4096 * 2);
    _Float16* c1t      = (_Float16*)carve(4096 * 2);
    _Float16* c2t      = (_Float16*)carve(4096 * 2);

    int n4 = (int)((off + 15) / 16);
    k_zero<<<(n4 + 255) / 256, 256, 0, stream>>>((float4*)d_ws, n4);
    k_initmax<<<(NB + 255) / 256, 256, 0, stream>>>(bmax);
    k_wconv<<<(4 * 4096 + 255) / 256, 256, 0, stream>>>(a_l1_w, a_l2_w, c_l1_w, c_l2_w,
                                                        w1t, w2t, c1t, c2t);
    k_req_encode<<<(N_REQ + 255) / 256, 256, 0, stream>>>(requests_x, rbatch,
        a_req_w, a_req_b, c_req_w, c_req_b, reqf16, reqmean, reqcnt);
    k_veh_encode<<<(N_VEH + 255) / 256, 256, 0, stream>>>(vehicles_x, vbatch,
        a_veh_w, a_veh_b, c_veh_w, c_veh_b, vehcat, cvehf, vehmean, vehcnt);
    k_v2p<<<(E_V2P + 255) / 256, 256, 0, stream>>>(passengers_x, v2p_s, v2p_r,
        a_pas_w, a_pas_b, c_pas_w, c_pas_b, passum_a, passum_c, pascnt);
    k_pasfin<<<(N_VEH + 255) / 256, 256, 0, stream>>>(passum_a, passum_c, pascnt,
        vehcat, cpasmean);
    k_r2v<<<(E_R2V + 255) / 256, 256, 0, stream>>>(r2v_s, r2v_r, vehcat, aggsum, aggcnt);
    k_actor_mlp<<<N_REQ / 16, 32, 0, stream>>>(reqf16, aggsum, aggcnt,
        w1t, a_l1_b, w2t, a_l2_b, a_l3_w, a_l3_b, rbatch, logits, bmax);
    k_sm2<<<(N_REQ + 255) / 256, 256, 0, stream>>>(logits, rbatch, bmax, bsum);
    k_sm3<<<(N_REQ + 255) / 256, 256, 0, stream>>>(logits, rbatch, bmax, bsum, bent);
    k_critic<<<NB / 16, 32, 0, stream>>>(reqmean, reqcnt, cvehf, cpasmean, vehmean, vehcnt,
        start_idx, c1t, c_l1_b, c2t, c_l2_b, c_l3_w, c_l3_b, out);
    k_final<<<(NB + 255) / 256, 256, 0, stream>>>(logits, rbatch, bmax, bsum,
        action_idx, bent, out);
}
